// TritonTriangleAttention_63797444215482
// MI455X (gfx1250) — matrix operand order repaired and checked by reading, not compiled, not run on hardware
//
#include <hip/hip_runtime.h>
#include <hip/hip_bf16.h>

// Problem constants: B=1, S=256, D=128, H=4, HD=32
#define S_LEN 256
#define D_DIM 128
#define NH    4
#define HDIM  32
#define QK_SCALE 0.17677669529663687f   // 1/sqrt(32)

typedef __attribute__((ext_vector_type(16))) _Float16 v16h;
typedef __attribute__((ext_vector_type(8)))  _Float16 v8h;
typedef __attribute__((ext_vector_type(4)))  _Float16 v4h;
typedef __attribute__((ext_vector_type(8)))  float    v8f;
typedef __attribute__((ext_vector_type(4)))  float    v4f;

__device__ __forceinline__ float bperm(float x, int srcLane) {
  return __builtin_bit_cast(float,
      __builtin_amdgcn_ds_bpermute(srcLane << 2, __builtin_bit_cast(int, x)));
}

// ---- WMMA fragment loaders (CDNA5 f16 16x16x32) ---------------------------
// A-matrix 16x32 (MxK): lane<16 -> row M=lane, K = {0..7, 16..23};
//                       lane>=16 -> row M=lane-16, K = {8..15, 24..31}.
__device__ __forceinline__ v16h load_fragA(const _Float16* __restrict__ base, int ld) {
  const int lane = threadIdx.x & 31;
  const int row  = lane & 15;
  const int kofs = (lane >> 4) << 3;          // 0 or 8
  const _Float16* p = base + (size_t)row * ld + kofs;
  v8h lo = *reinterpret_cast<const v8h*>(p);        // K +0..7
  v8h hi = *reinterpret_cast<const v8h*>(p + 16);   // K +16..23
  return __builtin_shufflevector(lo, hi, 0,1,2,3,4,5,6,7,8,9,10,11,12,13,14,15);
}

// B-matrix 32x16 (KxN): lane<16 -> col N=lane, K = {0..7, 16..23};
//                       lane>=16 -> col N=lane-16, K = {8..15, 24..31}.
// (same K order as the A fragment)
// For X @ W^T, B[k,n] = W[n,k]: lane loads two runs of 8 f16 of W row n (k+0..7, k+16..23).
__device__ __forceinline__ v16h load_fragB(const _Float16* __restrict__ base, int ld) {
  const int lane = threadIdx.x & 31;
  const int row  = lane & 15;
  const int kofs = (lane >> 4) << 3;          // 0 or 8
  const _Float16* p = base + (size_t)row * ld + kofs;
  v8h lo = *reinterpret_cast<const v8h*>(p);        // K +0..7
  v8h hi = *reinterpret_cast<const v8h*>(p + 16);   // K +16..23
  return __builtin_shufflevector(lo, hi, 0,1,2,3,4,5,6,7,8,9,10,11,12,13,14,15);
}

__device__ __forceinline__ v8f wmma16(v16h a, v16h b, v8f c) {
  return __builtin_amdgcn_wmma_f32_16x16x32_f16(false, a, false, b, (short)0, c,
                                                false, false);
}

// ---------------------------------------------------------------- kernel 0
// Convert the four 128x128 f32 weights to f16 (packed wq|wk|wv|wo).
__global__ void k_cvt_w(const float* __restrict__ wq, const float* __restrict__ wk,
                        const float* __restrict__ wv, const float* __restrict__ wo,
                        _Float16* __restrict__ w16) {
  int idx = blockIdx.x * blockDim.x + threadIdx.x;
  if (idx < D_DIM * D_DIM) {
    w16[idx]              = (_Float16)wq[idx];
    w16[16384 + idx]      = (_Float16)wk[idx];
    w16[2 * 16384 + idx]  = (_Float16)wv[idx];
    w16[3 * 16384 + idx]  = (_Float16)wo[idx];
  }
}

// ---------------------------------------------------------------- kernel 1
// LayerNorm over last dim (weight only). One wave per 128-elem row.
__global__ void k_ln(const float* __restrict__ pair, const float* __restrict__ lnw,
                     _Float16* __restrict__ xn) {
  const int lane = threadIdx.x & 31;
  const int wave = threadIdx.x >> 5;
  const int row  = blockIdx.x * 8 + wave;        // 0..65535
  const float* p = pair + (size_t)row * D_DIM + lane * 4;
  v4f x = *reinterpret_cast<const v4f*>(p);
  float s = x.x + x.y + x.z + x.w;
  #pragma unroll
  for (int m = 16; m >= 1; m >>= 1) s += bperm(s, lane ^ m);
  const float mu = s * (1.0f / 128.0f);
  v4f xc = x - mu;
  float s2 = xc.x * xc.x + xc.y * xc.y + xc.z * xc.z + xc.w * xc.w;
  #pragma unroll
  for (int m = 16; m >= 1; m >>= 1) s2 += bperm(s2, lane ^ m);
  const float rstd = rsqrtf(s2 * (1.0f / 128.0f) + 1e-5f);
  v4f w = *reinterpret_cast<const v4f*>(lnw + lane * 4);
  v4f y = xc * rstd * w;
  v4h h;
  h.x = (_Float16)y.x; h.y = (_Float16)y.y; h.z = (_Float16)y.z; h.w = (_Float16)y.w;
  *reinterpret_cast<v4h*>(xn + (size_t)row * D_DIM + lane * 4) = h;
}

// ---------------------------------------------------------------- kernel 2
// Q/K/V projections: xn @ W^T via WMMA. One wave per 16-row m-tile.
// Q is pre-scaled by 1/sqrt(HD). V is stored transposed per i: Vt[i][d][s].
__global__ void k_qkv(const _Float16* __restrict__ xn, const _Float16* __restrict__ w16,
                      _Float16* __restrict__ Qf, _Float16* __restrict__ Kf,
                      _Float16* __restrict__ Vt) {
  const int lane  = threadIdx.x & 31;
  const int t     = blockIdx.x * 8 + (threadIdx.x >> 5);  // m-tile 0..4095
  const int mbase = t * 16;
  const int i     = mbase >> 8;
  const int sbase = mbase & 255;
  const int col   = lane & 15;
  const int rofs  = (lane >> 4) << 3;

  // GL2 prefetch hint for the weight panels this wave will sweep.
  __builtin_prefetch(w16 + (size_t)lane * 1024, 0, 1);

  v16h a[4];
  #pragma unroll
  for (int kk = 0; kk < 4; ++kk)
    a[kk] = load_fragA(xn + (size_t)mbase * D_DIM + kk * 32, D_DIM);

  // Q then K (row-major outputs)
  #pragma unroll
  for (int wsel = 0; wsel < 2; ++wsel) {
    const _Float16* W = w16 + wsel * 16384;
    _Float16* out = wsel ? Kf : Qf;
    const float scl = wsel ? 1.0f : QK_SCALE;
    #pragma unroll
    for (int n0 = 0; n0 < 8; ++n0) {
      v8f acc = {};
      #pragma unroll
      for (int kk = 0; kk < 4; ++kk)
        acc = wmma16(a[kk], load_fragB(W + n0 * 16 * D_DIM + kk * 32, D_DIM), acc);
      _Float16* o = out + (size_t)mbase * D_DIM + n0 * 16 + col;
      #pragma unroll
      for (int v = 0; v < 8; ++v)
        o[(size_t)(v + rofs) * D_DIM] = (_Float16)(acc[v] * scl);
    }
  }
  // V, stored transposed: Vt[i*128*256 + d*256 + s]
  {
    const _Float16* W = w16 + 2 * 16384;
    #pragma unroll
    for (int n0 = 0; n0 < 8; ++n0) {
      v8f acc = {};
      #pragma unroll
      for (int kk = 0; kk < 4; ++kk)
        acc = wmma16(a[kk], load_fragB(W + n0 * 16 * D_DIM + kk * 32, D_DIM), acc);
      _Float16* o = Vt + (size_t)i * (D_DIM * S_LEN) + (size_t)(n0 * 16 + col) * S_LEN;
      #pragma unroll
      for (int v = 0; v < 8; ++v)
        o[sbase + v + rofs] = (_Float16)acc[v];
    }
  }
}

// ---------------------------------------------------------------- kernel 3
// Flash attention per (i, 16-query tile, head). One wave each; 4 waves/block.
// Trick: compute S^T = K_chunk x Q^T so the score D-layout (lane=query,
// vgpr=key) is exactly the A-operand layout needed for P x V — no shuffles.
__global__ void k_attn(const _Float16* __restrict__ Qf, const _Float16* __restrict__ Kf,
                       const _Float16* __restrict__ Vt, _Float16* __restrict__ At) {
  const int lane  = threadIdx.x & 31;
  const int h     = threadIdx.x >> 5;          // head 0..3
  const int i     = blockIdx.x >> 4;
  const int qbase = (blockIdx.x & 15) * 16;
  const int hi8   = (lane >> 4) << 3;
  const size_t ibase = (size_t)i * S_LEN * D_DIM;

  const _Float16* Qi  = Qf + ibase + (size_t)qbase * D_DIM + h * HDIM;
  const _Float16* Ki  = Kf + ibase + h * HDIM;
  const _Float16* Vti = Vt + (size_t)i * (D_DIM * S_LEN) + (size_t)(h * HDIM) * S_LEN;

  const v16h bq = load_fragB(Qi, D_DIM);       // B operand of S^T (lane = query)
  v8f O0 = {}, O1 = {};                        // out [16q x 32hd] as two D tiles
  float m = -1e30f, l = 0.0f;

  for (int c = 0; c < 8; ++c) {
    const int kb = c * 32;
    v16h a0 = load_fragA(Ki + (size_t)kb * D_DIM, D_DIM);         // keys kb..+15
    v16h a1 = load_fragA(Ki + (size_t)(kb + 16) * D_DIM, D_DIM);  // keys kb+16..+31
    v8f z = {};
    v8f s0 = wmma16(a0, bq, z);   // S^T tile, lane holds 8 scores of its query
    v8f s1 = wmma16(a1, bq, z);

    float lm = -1e30f;
    #pragma unroll
    for (int j = 0; j < 8; ++j) lm = fmaxf(lm, fmaxf(s0[j], s1[j]));
    lm = fmaxf(lm, bperm(lm, lane ^ 16));      // combine the query's two lanes
    const float mnew = fmaxf(m, lm);
    const float alpha = __expf(m - mnew);
    float ls = 0.0f;
    #pragma unroll
    for (int j = 0; j < 8; ++j) {
      s0[j] = __expf(s0[j] - mnew); ls += s0[j];
      s1[j] = __expf(s1[j] - mnew); ls += s1[j];
    }
    l = l * alpha + ls + bperm(ls, lane ^ 16);
    m = mnew;

    // rescale O: per-query alpha lives in lane (query); broadcast per vgpr row
    #pragma unroll
    for (int v = 0; v < 8; ++v) {
      float sc = bperm(alpha, v + hi8);
      O0[v] *= sc; O1[v] *= sc;
    }
    // P fragment: D-layout of S^T is already the WMMA A-layout for [16q x 32k]
    v16h p;
    #pragma unroll
    for (int j = 0; j < 8; ++j) {
      p[j]     = (_Float16)s0[j];
      p[8 + j] = (_Float16)s1[j];
    }
    v16h bv0 = load_fragB(Vti + kb, S_LEN);                // hd 0..15 (lane = hd)
    v16h bv1 = load_fragB(Vti + 16 * S_LEN + kb, S_LEN);   // hd 16..31
    O0 = wmma16(p, bv0, O0);
    O1 = wmma16(p, bv1, O1);
  }

  const float linv = 1.0f / l;
  _Float16* o = At + ibase + (size_t)qbase * D_DIM + h * HDIM + (lane & 15);
  #pragma unroll
  for (int v = 0; v < 8; ++v) {
    float sc = bperm(linv, v + hi8);
    o[(size_t)(v + hi8) * D_DIM]      = (_Float16)(O0[v] * sc);
    o[(size_t)(v + hi8) * D_DIM + 16] = (_Float16)(O1[v] * sc);
  }
}

// ---------------------------------------------------------------- kernel 4
// Output projection: attn @ wo^T -> f32. One wave per 16-row m-tile.
__global__ void k_oproj(const _Float16* __restrict__ At, const _Float16* __restrict__ w16,
                        float* __restrict__ out) {
  const int lane  = threadIdx.x & 31;
  const int t     = blockIdx.x * 8 + (threadIdx.x >> 5);
  const int mbase = t * 16;
  const int col   = lane & 15;
  const int rofs  = (lane >> 4) << 3;
  const _Float16* W = w16 + 3 * 16384;   // wo

  __builtin_prefetch(W + (size_t)lane * 512, 0, 1);

  v16h a[4];
  #pragma unroll
  for (int kk = 0; kk < 4; ++kk)
    a[kk] = load_fragA(At + (size_t)mbase * D_DIM + kk * 32, D_DIM);

  #pragma unroll
  for (int n0 = 0; n0 < 8; ++n0) {
    v8f acc = {};
    #pragma unroll
    for (int kk = 0; kk < 4; ++kk)
      acc = wmma16(a[kk], load_fragB(W + n0 * 16 * D_DIM + kk * 32, D_DIM), acc);
    float* o = out + (size_t)mbase * D_DIM + n0 * 16 + col;
    #pragma unroll
    for (int v = 0; v < 8; ++v)
      o[(size_t)(v + rofs) * D_DIM] = acc[v];
  }
}

// ---------------------------------------------------------------- launch
extern "C" void kernel_launch(void* const* d_in, const int* in_sizes, int n_in,
                              void* d_out, int out_size, void* d_ws, size_t ws_size,
                              hipStream_t stream) {
  const float* pair = (const float*)d_in[0];
  const float* lnw  = (const float*)d_in[1];
  const float* wq   = (const float*)d_in[2];
  const float* wk   = (const float*)d_in[3];
  const float* wv   = (const float*)d_in[4];
  const float* wo   = (const float*)d_in[5];
  float* out = (float*)d_out;

  char* ws = (char*)d_ws;
  const size_t MB16 = (size_t)1 << 24;               // 65536*128 f16 = 16 MiB
  _Float16* w16 = (_Float16*)(ws);                   // 128 KiB (4 weights f16)
  _Float16* xn  = (_Float16*)(ws + (1 << 17));
  _Float16* Qf  = (_Float16*)(ws + (1 << 17) + 1 * MB16);
  _Float16* Kf  = (_Float16*)(ws + (1 << 17) + 2 * MB16);
  _Float16* Vt  = (_Float16*)(ws + (1 << 17) + 3 * MB16);
  _Float16* At  = (_Float16*)(ws + (1 << 17) + 4 * MB16);

  k_cvt_w<<<64, 256, 0, stream>>>(wq, wk, wv, wo, w16);
  k_ln<<<8192, 256, 0, stream>>>(pair, lnw, xn);
  k_qkv<<<512, 256, 0, stream>>>(xn, w16, Qf, Kf, Vt);
  k_attn<<<4096, 128, 0, stream>>>(Qf, Kf, Vt, At);
  k_oproj<<<512, 256, 0, stream>>>(At, w16, out);
}
